// Decoder_74577812128590
// MI455X (gfx1250) — compile-verified, matrix operand order checked by reading, not run on hardware
//
#include <hip/hip_runtime.h>
#include <hip/hip_bf16.h>

// ---------------------------------------------------------------------------
// 2-layer LSTM greedy decoder for MI455X (gfx1250, wave32, WMMA).
//
// Shapes: B=4096, H=1024, C=2 classes, PRED_LEN=64 steps.
// Core op per step: gates(B x 4H) = h(B x H) @ W^T(H x 4H), fused with the
// LSTM cell update in the epilogue. bf16 WMMA (16x16x32) with f32 accum.
// The one-hot input GEMM is replaced by a gather of W_ih0[:, argmax].
//
// Wave tile: M=64 (4x 16-row tiles) x N=16 (j) x 4 gates = 16 WMMA / K-step,
// 16 b128 loads / K-step -> 1.0 loads per WMMA, 128 accum VGPRs.
// Block: 8 waves (4 M-waves x 2 N-waves) -> 256(b) x 32(j) tile.
// ---------------------------------------------------------------------------

#define HID 1024
#define BATCH 4096
#define NGATE 4
#define PRED_LEN 64
#define NCLS 2
#define MT 4                      // 16-row M tiles per wave (wave M = 64)

typedef __attribute__((ext_vector_type(16))) __bf16 v16bf;
typedef __attribute__((ext_vector_type(8)))  __bf16 v8bf;
typedef __attribute__((ext_vector_type(8)))  float  v8f;

// ---- fragment loader ------------------------------------------------------
// 16-bit A-matrix 16x32 layout (ISA 7.12.2): lane = (M row)%16, lane>>4 picks
// the K-half; per lane the 16 elements are two contiguous 8-element (16 byte)
// runs:  K = (h>>3)*16 + (lane>>4)*8 + (h&7).  B (K x N = W^T) is symmetric
// with lane = N column, same K formula -> contiguous along the weight row.
static __device__ __forceinline__ v16bf load_frag(const __bf16* __restrict__ rowp,
                                                  int k0, int hi) {
  v8bf lo = *(const v8bf*)(rowp + k0 + hi * 8);        // K = k0+hi*8 .. +7
  v8bf hp = *(const v8bf*)(rowp + k0 + 16 + hi * 8);   // K = k0+16+hi*8 .. +7
  return __builtin_shufflevector(lo, hp, 0,1,2,3,4,5,6,7,8,9,10,11,12,13,14,15);
}

static __device__ __forceinline__ float sigmoidf_fast(float x) {
  return 1.0f / (1.0f + __expf(-x));
}
static __device__ __forceinline__ float tanhf_fast(float x) {
  float e = __expf(2.0f * x);
  return (e - 1.0f) / (e + 1.0f);
}

// ---- fused LSTM layer ------------------------------------------------------
// gates[b][g*H+j] = A1[b,:]·W1[g*H+j,:] (+ A2[b,:]·W2[g*H+j,:]) + bsum[g*H+j]
//                   (+ Wx[g*H+j][xidx[b]] for layer 0)
// then c,h update fused in the epilogue.
__global__ __launch_bounds__(256)
void lstm_layer_kernel(const __bf16* __restrict__ A1, const __bf16* __restrict__ W1,
                       const __bf16* __restrict__ A2, const __bf16* __restrict__ W2,
                       const float*  __restrict__ bsum,
                       const float*  __restrict__ Wx,   // (4H,2) or null
                       const int*    __restrict__ xidx, // (B) or null
                       float* __restrict__ c,           // (B,H) in/out
                       __bf16* __restrict__ h_out)      // (B,H) out
{
  const int lane = threadIdx.x & 31;
  const int wid  = threadIdx.x >> 5;
  const int wm = wid >> 1, wn = wid & 1;
  const int n  = lane & 15, hi = lane >> 4;

  const int m0 = blockIdx.x * (128 * MT / 2) + wm * (16 * MT); // batch origin
  const int j0 = blockIdx.y * 32 + wn * 16;                    // j-tile origin
  const int j  = j0 + n;                                       // lane's column

  v8f acc[MT][NGATE] = {};                 // 128 accumulator VGPRs

  const int npass = (A2 != nullptr) ? 2 : 1;
  for (int pass = 0; pass < npass; ++pass) {
    const __bf16* A = pass ? A2 : A1;
    const __bf16* W = pass ? W2 : W1;

    const __bf16* arow[MT];
#pragma unroll
    for (int mi = 0; mi < MT; ++mi)
      arow[mi] = A + (size_t)(m0 + mi * 16 + n) * HID;
    const __bf16* wrow[NGATE];
#pragma unroll
    for (int g = 0; g < NGATE; ++g)
      wrow[g] = W + (size_t)(g * HID + j) * HID;

    for (int k0 = 0; k0 < HID; k0 += 32) {
      if (k0 + 64 < HID) {
#pragma unroll
        for (int g = 0; g < NGATE; ++g)          // -> global_prefetch_b8
          __builtin_prefetch(wrow[g] + k0 + 64, 0, 1);
      }
      v16bf bfrg[NGATE];
#pragma unroll
      for (int g = 0; g < NGATE; ++g) bfrg[g] = load_frag(wrow[g], k0, hi);
      v16bf af[MT];
#pragma unroll
      for (int mi = 0; mi < MT; ++mi) af[mi] = load_frag(arow[mi], k0, hi);

#pragma unroll
      for (int mi = 0; mi < MT; ++mi)
#pragma unroll
        for (int g = 0; g < NGATE; ++g)
          acc[mi][g] = __builtin_amdgcn_wmma_f32_16x16x32_bf16(
              false, af[mi], false, bfrg[g], (short)0, acc[mi][g], false, false);
    }
  }

  // ---- fused LSTM cell epilogue (torch gate order: i, f, g, o) ----
  const float bi  = bsum[0 * HID + j];
  const float bff = bsum[1 * HID + j];
  const float bgg = bsum[2 * HID + j];
  const float boo = bsum[3 * HID + j];

#pragma unroll
  for (int mi = 0; mi < MT; ++mi) {
    const int mbase = m0 + mi * 16 + hi * 8;     // D layout: m = r + (lane>>4)*8
#pragma unroll
    for (int r = 0; r < 8; ++r) {
      const int b = mbase + r;
      float ig = acc[mi][0][r] + bi;
      float fg = acc[mi][1][r] + bff;
      float gg = acc[mi][2][r] + bgg;
      float og = acc[mi][3][r] + boo;
      if (Wx) {                                  // one-hot input term (layer 0)
        int xi = xidx[b];
        if (xi >= 0) {
          ig += Wx[(0 * HID + j) * 2 + xi];
          fg += Wx[(1 * HID + j) * 2 + xi];
          gg += Wx[(2 * HID + j) * 2 + xi];
          og += Wx[(3 * HID + j) * 2 + xi];
        }
      }
      const size_t idx = (size_t)b * HID + j;
      float cn = sigmoidf_fast(fg) * c[idx] + sigmoidf_fast(ig) * tanhf_fast(gg);
      c[idx] = cn;
      h_out[idx] = (__bf16)(sigmoidf_fast(og) * tanhf_fast(cn));
    }
  }
}

// ---- FC head (2 classes) + greedy argmax feedback --------------------------
__global__ __launch_bounds__(256)
void logits_argmax_kernel(const __bf16* __restrict__ h1, const float* __restrict__ fcW,
                          const float* __restrict__ fcb, float* __restrict__ out,
                          int* __restrict__ xidx, int t)
{
  const int lane = threadIdx.x & 31;
  const int wid  = threadIdx.x >> 5;
  const int b = blockIdx.x * 8 + wid;            // one wave per batch row
  const __bf16* hrow = h1 + (size_t)b * HID;
  float a0 = 0.f, a1 = 0.f;
  for (int k = lane; k < HID; k += 32) {
    float hv = (float)hrow[k];
    a0 += hv * fcW[k];
    a1 += hv * fcW[HID + k];
  }
#pragma unroll
  for (int off = 16; off; off >>= 1) {
    a0 += __shfl_xor(a0, off, 32);
    a1 += __shfl_xor(a1, off, 32);
  }
  if (lane == 0) {
    float l0 = a0 + fcb[0], l1 = a1 + fcb[1];
    size_t o = ((size_t)b * PRED_LEN + t) * NCLS;
    out[o] = l0;
    out[o + 1] = l1;
    xidx[b] = (l1 > l0) ? 1 : 0;                 // argmax, ties -> class 0
  }
}

// ---- prep kernels -----------------------------------------------------------
__global__ void f32_to_bf16_kernel(const float* __restrict__ s, __bf16* __restrict__ d, int n) {
  int i = blockIdx.x * blockDim.x + threadIdx.x;
  if (i < n) d[i] = (__bf16)s[i];
}
__global__ void bias_sum_kernel(const float* __restrict__ a, const float* __restrict__ b,
                                float* __restrict__ o, int n) {
  int i = blockIdx.x * blockDim.x + threadIdx.x;
  if (i < n) o[i] = a[i] + b[i];
}
__global__ void init_xidx_kernel(int* __restrict__ x, int n) {
  int i = blockIdx.x * blockDim.x + threadIdx.x;
  if (i < n) x[i] = -1;                          // t=0: x is all-zeros, no Wx term
}

// ---- host orchestration -----------------------------------------------------
extern "C" void kernel_launch(void* const* d_in, const int* in_sizes, int n_in,
                              void* d_out, int out_size, void* d_ws, size_t ws_size,
                              hipStream_t stream) {
  (void)in_sizes; (void)n_in; (void)out_size; (void)ws_size;

  const float* h_in  = (const float*)d_in[0];   // (2,B,H)
  const float* c_in  = (const float*)d_in[1];   // (2,B,H)
  const float* W_ih0 = (const float*)d_in[2];   // (4H,2)
  const float* W_hh0 = (const float*)d_in[3];   // (4H,H)
  const float* b_ih0 = (const float*)d_in[4];
  const float* b_hh0 = (const float*)d_in[5];
  const float* W_ih1 = (const float*)d_in[6];   // (4H,H)
  const float* W_hh1 = (const float*)d_in[7];   // (4H,H)
  const float* b_ih1 = (const float*)d_in[8];
  const float* b_hh1 = (const float*)d_in[9];
  const float* fc_W  = (const float*)d_in[10];  // (2,H)
  const float* fc_b  = (const float*)d_in[11];

  // workspace carve-out (~92 MB)
  char* ws = (char*)d_ws;
  size_t off = 0;
  auto take = [&](size_t bytes) -> char* {
    char* p = ws + off;
    off = (off + bytes + 255) & ~(size_t)255;
    return p;
  };
  const size_t WN = (size_t)NGATE * HID * HID;          // 4M elems per weight
  const size_t SN = (size_t)BATCH * HID;                // 4M elems per state
  __bf16* Wb_hh0 = (__bf16*)take(WN * sizeof(__bf16));
  __bf16* Wb_ih1 = (__bf16*)take(WN * sizeof(__bf16));
  __bf16* Wb_hh1 = (__bf16*)take(WN * sizeof(__bf16));
  __bf16* hb0    = (__bf16*)take(2 * SN * sizeof(__bf16));   // double buffer
  __bf16* hb1    = (__bf16*)take(2 * SN * sizeof(__bf16));   // double buffer
  float*  cbuf   = (float*)take(2 * SN * sizeof(float));     // c0 | c1
  float*  bsum0  = (float*)take((size_t)NGATE * HID * sizeof(float));
  float*  bsum1  = (float*)take((size_t)NGATE * HID * sizeof(float));
  int*    xidx   = (int*)take((size_t)BATCH * sizeof(int));

  float* c0 = cbuf;
  float* c1 = cbuf + SN;

  const int thr = 256;
  auto cdiv = [](size_t a, int b) { return (unsigned)((a + b - 1) / b); };

  // prep: bf16 weights/state, summed biases, in-place c state, x=zeros
  f32_to_bf16_kernel<<<cdiv(WN, thr), thr, 0, stream>>>(W_hh0, Wb_hh0, (int)WN);
  f32_to_bf16_kernel<<<cdiv(WN, thr), thr, 0, stream>>>(W_ih1, Wb_ih1, (int)WN);
  f32_to_bf16_kernel<<<cdiv(WN, thr), thr, 0, stream>>>(W_hh1, Wb_hh1, (int)WN);
  f32_to_bf16_kernel<<<cdiv(SN, thr), thr, 0, stream>>>(h_in,      hb0, (int)SN);
  f32_to_bf16_kernel<<<cdiv(SN, thr), thr, 0, stream>>>(h_in + SN, hb1, (int)SN);
  hipMemcpyAsync(cbuf, c_in, 2 * SN * sizeof(float), hipMemcpyDeviceToDevice, stream);
  bias_sum_kernel<<<cdiv(NGATE * HID, thr), thr, 0, stream>>>(b_ih0, b_hh0, bsum0, NGATE * HID);
  bias_sum_kernel<<<cdiv(NGATE * HID, thr), thr, 0, stream>>>(b_ih1, b_hh1, bsum1, NGATE * HID);
  init_xidx_kernel<<<cdiv(BATCH, thr), thr, 0, stream>>>(xidx, BATCH);

  dim3 ggrid(BATCH / (64 * MT), HID / 32);   // 16 x 32 blocks, 8 waves each
  for (int t = 0; t < PRED_LEN; ++t) {
    const __bf16* h0c = hb0 + (size_t)(t & 1) * SN;
    __bf16*       h0n = hb0 + (size_t)((t + 1) & 1) * SN;
    const __bf16* h1c = hb1 + (size_t)(t & 1) * SN;
    __bf16*       h1n = hb1 + (size_t)((t + 1) & 1) * SN;

    // layer 0: gates = h0·W_hh0^T + (b_ih0+b_hh0) + W_ih0[:,x]
    lstm_layer_kernel<<<ggrid, thr, 0, stream>>>(
        h0c, Wb_hh0, nullptr, nullptr, bsum0, W_ih0, xidx, c0, h0n);
    // layer 1: gates = h0n·W_ih1^T + h1·W_hh1^T + (b_ih1+b_hh1)
    lstm_layer_kernel<<<ggrid, thr, 0, stream>>>(
        h0n, Wb_ih1, h1c, Wb_hh1, bsum1, nullptr, nullptr, c1, h1n);
    // logits + greedy argmax feedback
    logits_argmax_kernel<<<BATCH / 8, thr, 0, stream>>>(
        h1n, fc_W, fc_b, (float*)d_out, xidx, t);
  }
}
